// GRUEmbedder_84610855731264
// MI455X (gfx1250) — matrix operand order repaired and checked by reading, not compiled, not run on hardware
//
#include <hip/hip_runtime.h>
#include <hip/hip_bf16.h>
#include <math.h>

// GRU final-hidden-state kernel for MI455X (gfx1250, wave32, WMMA).
// Phase 0: pad/convert weights to f16 (W_ihP 384x96 zero-padded, W_hhP 384x128).
// Phase 1: gx = x^T @ W_ih^T + b_ih  (WMMA f16, K padded 80->96), f16 in d_ws.
// Phase 2: per-batch-tile recurrence, W_hh held in VGPRs as f16 B-fragments,
//          h staged through LDS, 12 v_wmma_f32_16x16x32_f16 per wave per step.

typedef _Float16 v16h __attribute__((ext_vector_type(16)));
typedef _Float16 v8h  __attribute__((ext_vector_type(8)));
typedef float    v8f  __attribute__((ext_vector_type(8)));

#define WMMA_F16(a, b, c) \
  __builtin_amdgcn_wmma_f32_16x16x32_f16(false, (a), false, (b), (short)0, (c), false, false)

#define GRU_B 256
#define GRU_M 80
#define GRU_KP 96   // M padded to 3 k-chunks of 32
#define GRU_T 1000
#define GRU_H 128
#define GRU_G 384   // 3*H

union AFrag { v16h v; v8h h[2]; };

// ---------------------------------------------------------------------------
// Kernel 0: weight pad/convert. W_ihP[g][k] = k<80 ? W_ih[g][k] : 0 (f16),
// W_hhP = f16(W_hh). One-shot, 192 blocks x 256 threads.
// ---------------------------------------------------------------------------
__global__ __launch_bounds__(256)
void gru_prep_kernel(const float* __restrict__ W_ih,
                     const float* __restrict__ W_hh,
                     _Float16* __restrict__ W_ihP,
                     _Float16* __restrict__ W_hhP)
{
  int i = blockIdx.x * 256 + threadIdx.x;
  if (i < GRU_G * GRU_KP) {
    int g = i / GRU_KP, k = i - g * GRU_KP;
    W_ihP[i] = (k < GRU_M) ? (_Float16)W_ih[g * GRU_M + k] : (_Float16)0.0f;
  }
  if (i < GRU_G * GRU_H) {
    W_hhP[i] = (_Float16)W_hh[i];
  }
}

// ---------------------------------------------------------------------------
// Kernel 1: gx[bt][g] = sum_m x[b][m][t] * W_ih[g][m] + b_ih[g]   (f16 out)
// Grid: (B*T)/16 = 16000 blocks, 128 threads (4 waves). Each block does one
// 16-row strip of the (256000 x 384) output; each wave owns 6 N-tiles.
// ---------------------------------------------------------------------------
__global__ __launch_bounds__(128)
void gru_gx_kernel(const float* __restrict__ x,
                   const _Float16* __restrict__ W_ihP,
                   const float* __restrict__ b_ih,
                   _Float16* __restrict__ gx)
{
  __shared__ __align__(16) _Float16 xA[16 * GRU_KP];  // 16 rows x K(padded 96)

  const int tid  = threadIdx.x;
  const int lane = tid & 31;
  const int wave = tid >> 5;          // 0..3
  const int base = blockIdx.x * 16;   // bt base

  // Stage A tile into LDS (branch-free): valid region then pad region.
  for (int idx = tid; idx < 16 * GRU_M; idx += 128) {
    int r = idx / GRU_M, k = idx - r * GRU_M;
    int bt = base + r;
    int b  = bt / GRU_T;
    int t  = bt - b * GRU_T;
    xA[r * GRU_KP + k] = (_Float16)x[((size_t)b * GRU_M + k) * GRU_T + t];
  }
  for (int idx = tid; idx < 16 * (GRU_KP - GRU_M); idx += 128) {
    int r = idx >> 4, k = GRU_M + (idx & 15);
    xA[r * GRU_KP + k] = (_Float16)0.0f;
  }
  __syncthreads();

  const int m  = lane & 15;   // A row (and B column) index for this lane
  const int hl = lane >> 4;   // half-lane select

  // A fragments: 16x32 f16 layout — lane(m,hl), halves K = kc*32 + 16*hl + 0..15.
  v16h a[3];
#pragma unroll
  for (int kc = 0; kc < 3; ++kc) {
    const v8h* p = (const v8h*)&xA[m * GRU_KP + kc * 32 + 16 * hl];
    AFrag u; u.h[0] = p[0]; u.h[1] = p[1];  // +8 halves
    a[kc] = u.v;
  }

#pragma unroll
  for (int i = 0; i < 6; ++i) {
    int nt = wave * 6 + i;        // 0..23
    int g  = nt * 16 + m;         // output gate column for this lane
    // B fragments: 32x16 f16 — lane = N, halves K = kc*32 + 16*hl + e.
    // Pre-padded f16 weights: one aligned 32B contiguous load per fragment.
    v8f c = {};
#pragma unroll
    for (int kc = 0; kc < 3; ++kc) {
      int kb = kc * 32 + 16 * hl;
      v16h bb = *(const v16h*)(W_ihP + g * GRU_KP + kb);
      c = WMMA_F16(a[kc], bb, c);
    }
    float bias = b_ih[g];
    // C layout: VGPR r -> row m = r + 8*hl, col = g.
#pragma unroll
    for (int r = 0; r < 8; ++r) {
      int bt = base + r + 8 * hl;
      gx[(size_t)bt * GRU_G + g] = (_Float16)(c[r] + bias);
    }
  }
}

// ---------------------------------------------------------------------------
// Kernel 2: per-16-batch-row GRU recurrence. Grid: 16 blocks x 256 threads
// (8 waves). Wave w owns gate N-tiles {3w,3w+1,3w+2}; its 12 W_hh B-fragments
// (3 tiles x 4 k-chunks) live in VGPRs for all 1000 steps.
// ---------------------------------------------------------------------------
__global__ __launch_bounds__(256)
void gru_rec_kernel(const _Float16* __restrict__ gx,
                    const _Float16* __restrict__ W_hhP,
                    const float* __restrict__ b_hh,
                    float* __restrict__ out)
{
  __shared__ __align__(16) _Float16 hA[16 * GRU_H];   // h as f16, A staging (4 KB)
  __shared__ __align__(16) float    ghS[16 * GRU_G];  // gh accumulators (24 KB)

  const int tid   = threadIdx.x;
  const int lane  = tid & 31;
  const int wave  = tid >> 5;          // 0..7
  const int bbase = blockIdx.x * 16;   // batch rows [bbase, bbase+16)

  const int m  = lane & 15;
  const int hl = lane >> 4;

  // Preload W_hh B-fragments from pre-converted f16: one v16h load each.
  v16h bfr[3][4];
#pragma unroll
  for (int i = 0; i < 3; ++i) {
    int g = (wave * 3 + i) * 16 + m;
#pragma unroll
    for (int kc = 0; kc < 4; ++kc) {
      int kb = kc * 32 + 16 * hl;
      bfr[i][kc] = *(const v16h*)(W_hhP + g * GRU_H + kb);
    }
  }

  // Elementwise ownership: thread -> (row em, hidden units jb..jb+7).
  const int em = tid >> 4;
  const int jb = (tid & 15) * 8;
  float hreg[8], br[8], bz[8], bn[8];
#pragma unroll
  for (int q = 0; q < 8; ++q) {
    hreg[q] = 0.0f;
    br[q] = b_hh[jb + q];
    bz[q] = b_hh[GRU_H + jb + q];
    bn[q] = b_hh[2 * GRU_H + jb + q];
  }
  for (int idx = tid; idx < 16 * GRU_H; idx += 256) hA[idx] = (_Float16)0.0f;
  __syncthreads();

  for (int t = 0; t < GRU_T; ++t) {
    // ---- GEMM phase: gh = h @ W_hh^T (16 x 384), h read from LDS as f16 ----
    v16h a[4];
#pragma unroll
    for (int kc = 0; kc < 4; ++kc) {
      const v8h* p = (const v8h*)&hA[m * GRU_H + kc * 32 + 16 * hl];
      AFrag u; u.h[0] = p[0]; u.h[1] = p[1];
      a[kc] = u.v;
    }
    // Hide next step's gx latency behind WMMA issue.
    const size_t gxrow = ((size_t)(bbase + em) * GRU_T + t) * GRU_G;
    if (t + 1 < GRU_T) {
      __builtin_prefetch(&gx[gxrow + GRU_G + jb], 0, 1);
      __builtin_prefetch(&gx[gxrow + GRU_G + GRU_H + jb], 0, 1);
      __builtin_prefetch(&gx[gxrow + GRU_G + 2 * GRU_H + jb], 0, 1);
    }
#pragma unroll
    for (int i = 0; i < 3; ++i) {
      v8f c = {};
#pragma unroll
      for (int kc = 0; kc < 4; ++kc)
        c = WMMA_F16(a[kc], bfr[i][kc], c);
      int g = (wave * 3 + i) * 16 + m;
#pragma unroll
      for (int r = 0; r < 8; ++r)
        ghS[(r + 8 * hl) * GRU_G + g] = c[r];
    }
    __syncthreads();

    // ---- Elementwise gate phase ----
    const v8h* gp = (const v8h*)(gx + gxrow);
    v8h xrv = gp[(tid & 15)];
    v8h xzv = gp[16 + (tid & 15)];
    v8h xnv = gp[32 + (tid & 15)];
    v8h hv;
#pragma unroll
    for (int q = 0; q < 8; ++q) {
      int j = jb + q;
      float hr = ghS[em * GRU_G + j] + br[q];
      float hz = ghS[em * GRU_G + GRU_H + j] + bz[q];
      float hn = ghS[em * GRU_G + 2 * GRU_H + j] + bn[q];
      float r = 1.0f / (1.0f + __expf(-((float)xrv[q] + hr)));
      float z = 1.0f / (1.0f + __expf(-((float)xzv[q] + hz)));
      float n = tanhf((float)xnv[q] + r * hn);
      float h = (1.0f - z) * n + z * hreg[q];
      hreg[q] = h;
      hv[q] = (_Float16)h;
    }
    *(v8h*)&hA[em * GRU_H + jb] = hv;
    __syncthreads();
  }

#pragma unroll
  for (int q = 0; q < 8; ++q)
    out[(size_t)(bbase + em) * GRU_H + jb + q] = hreg[q];
}

// ---------------------------------------------------------------------------
extern "C" void kernel_launch(void* const* d_in, const int* in_sizes, int n_in,
                              void* d_out, int out_size, void* d_ws, size_t ws_size,
                              hipStream_t stream) {
  const float* x    = (const float*)d_in[0];  // (B, M, T)
  const float* W_ih = (const float*)d_in[1];  // (3H, M)
  const float* W_hh = (const float*)d_in[2];  // (3H, H)
  const float* b_ih = (const float*)d_in[3];  // (3H)
  const float* b_hh = (const float*)d_in[4];  // (3H)
  float* out = (float*)d_out;                 // (B, H)

  // Workspace layout (f16):
  //   gx:    (B*T) x 3H           = 256000*384*2 bytes (~197 MB)
  //   W_ihP: 384 x 96  (padded)
  //   W_hhP: 384 x 128
  char* ws = (char*)d_ws;
  _Float16* gx = (_Float16*)ws;
  size_t gx_bytes = (size_t)GRU_B * GRU_T * GRU_G * sizeof(_Float16);
  gx_bytes = (gx_bytes + 255) & ~(size_t)255;
  _Float16* W_ihP = (_Float16*)(ws + gx_bytes);
  _Float16* W_hhP = W_ihP + (size_t)GRU_G * GRU_KP;

  gru_prep_kernel<<<(GRU_G * GRU_H + 255) / 256, 256, 0, stream>>>(W_ih, W_hh, W_ihP, W_hhP);
  gru_gx_kernel<<<(GRU_B * GRU_T) / 16, 128, 0, stream>>>(x, W_ihP, b_ih, gx);
  gru_rec_kernel<<<GRU_B / 16, 256, 0, stream>>>(gx, W_hhP, b_hh, out);
}
